// Pointnet2MSGBackbone_52896817217694
// MI455X (gfx1250) — matrix operand order repaired and checked by reading, not compiled, not run on hardware
//
#include <hip/hip_runtime.h>
#include <hip/hip_bf16.h>

// ---------------------------------------------------------------------------
// PointNet++ MSG backbone for gfx1250 (MI455X).
// Dense math (all pointwise MLP layers) runs on v_wmma_f32_16x16x32_bf16,
// register-blocked 2x2 (32x32 output tile per wave) for 2 loads per WMMA.
// ---------------------------------------------------------------------------

typedef __attribute__((ext_vector_type(16))) __bf16 v16bf;
typedef __attribute__((ext_vector_type(8)))  float  v8f;

__device__ __forceinline__ unsigned short f2bf(float f) {
    unsigned int u = __float_as_uint(f);
    // round-to-nearest-even bf16
    unsigned int r = (u + 0x7FFFu + ((u >> 16) & 1u)) >> 16;
    return (unsigned short)r;
}

// ---------------------------------------------------------------------------
// Farthest point sampling: one persistent block per batch, LDS argmax.
// ---------------------------------------------------------------------------
__global__ __launch_bounds__(1024)
void fps_kernel(const float* __restrict__ xyz, int N, int npoint,
                int* __restrict__ idx_out, float* __restrict__ dist) {
    const int b   = blockIdx.x;
    const int tid = threadIdx.x;
    const int nt  = blockDim.x;
    const float* px = xyz + (size_t)b * N * 3;
    float* pd = dist + (size_t)b * N;

    __shared__ float sv[1024];
    __shared__ int   si[1024];
    __shared__ int   s_far;
    if (tid == 0) s_far = 0;
    __syncthreads();

    for (int m = 0; m < npoint; ++m) {
        const int far = s_far;
        if (tid == 0) idx_out[(size_t)b * npoint + m] = far;
        const float cx = px[(size_t)far * 3 + 0];
        const float cy = px[(size_t)far * 3 + 1];
        const float cz = px[(size_t)far * 3 + 2];
        float bv = -1.0f; int bi = 0;
        for (int i = tid; i < N; i += nt) {
            const float dx = px[(size_t)i * 3 + 0] - cx;
            const float dy = px[(size_t)i * 3 + 1] - cy;
            const float dz = px[(size_t)i * 3 + 2] - cz;
            const float d  = dx * dx + dy * dy + dz * dz;
            const float pv = (m == 0) ? 1e10f : pd[i];
            const float nd = fminf(pv, d);
            pd[i] = nd;
            if (nd > bv) { bv = nd; bi = i; }
        }
        sv[tid] = bv; si[tid] = bi;
        __syncthreads();
        for (int s = nt >> 1; s > 0; s >>= 1) {
            if (tid < s) {
                if (sv[tid + s] > sv[tid] ||
                    (sv[tid + s] == sv[tid] && si[tid + s] < si[tid])) {
                    sv[tid] = sv[tid + s]; si[tid] = si[tid + s];
                }
            }
            __syncthreads();
        }
        if (tid == 0) s_far = si[0];
        __syncthreads();
    }
}

__global__ void gather_xyz_kernel(const float* __restrict__ xyz,
                                  const int* __restrict__ idx,
                                  float* __restrict__ out,
                                  int np, int N, int total) {
    int t = blockIdx.x * blockDim.x + threadIdx.x;
    if (t >= total) return;
    int b = t / np;
    int g = idx[t];
    const float* p = xyz + ((size_t)b * N + g) * 3;
    float* o = out + (size_t)t * 3;
    o[0] = p[0]; o[1] = p[1]; o[2] = p[2];
}

// ---------------------------------------------------------------------------
// Ball query: first NSAMPLE indices (ascending) within radius, pad with first.
// ---------------------------------------------------------------------------
__global__ void ball_query_kernel(const float* __restrict__ xyz,
                                  const float* __restrict__ new_xyz,
                                  int N, int np, float r2,
                                  int* __restrict__ gi, int total) {
    int t = blockIdx.x * blockDim.x + threadIdx.x;
    if (t >= total) return;
    int b = t / np;
    const float* q  = new_xyz + (size_t)t * 3;
    const float* px = xyz + (size_t)b * N * 3;
    int* out = gi + (size_t)t * 32;
    const float qx = q[0], qy = q[1], qz = q[2];
    int cnt = 0, first = 0; bool hasfirst = false;
    for (int i = 0; i < N && cnt < 32; ++i) {
        const float dx = px[(size_t)i * 3 + 0] - qx;
        const float dy = px[(size_t)i * 3 + 1] - qy;
        const float dz = px[(size_t)i * 3 + 2] - qz;
        if (dx * dx + dy * dy + dz * dz < r2) {
            if (!hasfirst) { first = i; hasfirst = true; }
            out[cnt++] = i;
        }
    }
    for (; cnt < 32; ++cnt) out[cnt] = first;
}

// ---------------------------------------------------------------------------
// Grouping: A[row][k] = bf16( [xyz(gi)-new_xyz | feats(gi) | 0-pad] )
// ---------------------------------------------------------------------------
__global__ void group_kernel(const float* __restrict__ xyz,
                             const float* __restrict__ feats,
                             const float* __restrict__ new_xyz,
                             const int* __restrict__ gi,
                             unsigned short* __restrict__ A,
                             int np, int N, int Cf, int Kpad, int total) {
    int t = blockIdx.x * blockDim.x + threadIdx.x;     // (b, s, k) flattened
    if (t >= total) return;
    int b   = t / (np * 32);
    int rem = t % (np * 32);
    int s   = rem / 32;
    int g   = gi[t];
    const float* p = xyz + ((size_t)b * N + g) * 3;
    const float* q = new_xyz + ((size_t)b * np + s) * 3;
    unsigned short* row = A + (size_t)t * Kpad;
    row[0] = f2bf(p[0] - q[0]);
    row[1] = f2bf(p[1] - q[1]);
    row[2] = f2bf(p[2] - q[2]);
    if (Cf > 0) {
        const float* f = feats + ((size_t)b * N + g) * Cf;
        for (int c = 0; c < Cf; ++c) row[3 + c] = f2bf(f[c]);
    }
    for (int c = 3 + Cf; c < Kpad; ++c) row[c] = 0;
}

// ---------------------------------------------------------------------------
// Weight transpose+convert: W[ci][co] f32 -> Wt[co][Kpad] bf16 (K zero-pad)
// ---------------------------------------------------------------------------
__global__ void cvt_weight_kernel(const float* __restrict__ W,
                                  unsigned short* __restrict__ Wt,
                                  int ci, int co, int Kp, int total) {
    int t = blockIdx.x * blockDim.x + threadIdx.x;
    if (t >= total) return;
    int n = t / Kp, k = t % Kp;
    float v = (k < ci) ? W[(size_t)k * co + n] : 0.0f;
    Wt[t] = f2bf(v);
}

__global__ void cvt_act_kernel(const float* __restrict__ src,
                               unsigned short* __restrict__ dst, size_t n) {
    size_t t = (size_t)blockIdx.x * blockDim.x + threadIdx.x;
    if (t < n) dst[t] = f2bf(src[t]);
}

// ---------------------------------------------------------------------------
// WMMA GEMM:  out[M][N] = relu(A[M][K](bf16) * Wt[N][K]^T(bf16) + bias[N])
// 32x32 output tile per wave (2x2 register blocking): per K-step of 32 we do
// 8 global_load_b128 feeding 4 v_wmma_f32_16x16x32_bf16 (2 loads per WMMA).
// Fragment lane layouts per CDNA5 ISA 7.12.2:
//   A (16x32 bf16): lane r+16h holds row M=r; element e holds
//                   K = (e<8?0:16) + 8h + (e&7)  -> two 16B loads per lane.
//   B (32x16 bf16): lane r+16h holds col N=r; element e holds
//                   K = (e<8?0:16) + 8h + (e&7)  -> two 16B loads per lane (W pre-transposed).
//   C/D (16x16 f32): element j, lane r+16h -> M = j + 8h, N = r.
// ---------------------------------------------------------------------------
__global__ __launch_bounds__(128)
void gemm_bias_relu_kernel(const unsigned short* __restrict__ A,
                           const unsigned short* __restrict__ Wt,
                           const float* __restrict__ bias,
                           float* __restrict__ out,
                           int Mrows, int Ncols, int K) {
    const int wave = threadIdx.y;                  // 0..3
    const int lane = threadIdx.x;                  // 0..31
    const int m0 = (blockIdx.x * 4 + wave) * 32;
    const int n0 = blockIdx.y * 32;
    if (m0 >= Mrows) return;
    const int h = lane >> 4;
    const int r = lane & 15;

    const unsigned short* arow0 = A  + (size_t)(m0 + r) * K;
    const unsigned short* arow1 = A  + (size_t)(m0 + 16 + r) * K;
    const unsigned short* wrow0 = Wt + (size_t)(n0 + r) * K;
    const unsigned short* wrow1 = Wt + (size_t)(n0 + 16 + r) * K;

    union Frag { v16bf v; uint4 q[2]; };
    v8f acc00 = {0.f, 0.f, 0.f, 0.f, 0.f, 0.f, 0.f, 0.f};
    v8f acc01 = acc00, acc10 = acc00, acc11 = acc00;

    for (int k0 = 0; k0 < K; k0 += 32) {
        Frag a0, a1, b0, b1;
        a0.q[0] = *reinterpret_cast<const uint4*>(arow0 + k0 + h * 8);
        a0.q[1] = *reinterpret_cast<const uint4*>(arow0 + k0 + 16 + h * 8);
        a1.q[0] = *reinterpret_cast<const uint4*>(arow1 + k0 + h * 8);
        a1.q[1] = *reinterpret_cast<const uint4*>(arow1 + k0 + 16 + h * 8);
        b0.q[0] = *reinterpret_cast<const uint4*>(wrow0 + k0 + h * 8);
        b0.q[1] = *reinterpret_cast<const uint4*>(wrow0 + k0 + 16 + h * 8);
        b1.q[0] = *reinterpret_cast<const uint4*>(wrow1 + k0 + h * 8);
        b1.q[1] = *reinterpret_cast<const uint4*>(wrow1 + k0 + 16 + h * 8);
        __builtin_prefetch(arow0 + k0 + 32, 0, 3);   // global_prefetch_b8 (near)
        __builtin_prefetch(arow1 + k0 + 32, 0, 3);
        acc00 = __builtin_amdgcn_wmma_f32_16x16x32_bf16(
                    false, a0.v, false, b0.v, (short)0, acc00, false, false);
        acc01 = __builtin_amdgcn_wmma_f32_16x16x32_bf16(
                    false, a0.v, false, b1.v, (short)0, acc01, false, false);
        acc10 = __builtin_amdgcn_wmma_f32_16x16x32_bf16(
                    false, a1.v, false, b0.v, (short)0, acc10, false, false);
        acc11 = __builtin_amdgcn_wmma_f32_16x16x32_bf16(
                    false, a1.v, false, b1.v, (short)0, acc11, false, false);
    }

    const int col0 = n0 + r;
    const int col1 = n0 + 16 + r;
    const float bv0 = (col0 < Ncols) ? bias[col0] : 0.0f;
    const float bv1 = (col1 < Ncols) ? bias[col1] : 0.0f;
#pragma unroll
    for (int j = 0; j < 8; ++j) {
        const int rm0 = m0 + j + 8 * h;
        const int rm1 = m0 + 16 + j + 8 * h;
        if (rm0 < Mrows) {
            if (col0 < Ncols) {
                float v = acc00[j] + bv0;
                out[(size_t)rm0 * Ncols + col0] = v > 0.0f ? v : 0.0f;
            }
            if (col1 < Ncols) {
                float v = acc01[j] + bv1;
                out[(size_t)rm0 * Ncols + col1] = v > 0.0f ? v : 0.0f;
            }
        }
        if (rm1 < Mrows) {
            if (col0 < Ncols) {
                float v = acc10[j] + bv0;
                out[(size_t)rm1 * Ncols + col0] = v > 0.0f ? v : 0.0f;
            }
            if (col1 < Ncols) {
                float v = acc11[j] + bv1;
                out[(size_t)rm1 * Ncols + col1] = v > 0.0f ? v : 0.0f;
            }
        }
    }
}

// ---------------------------------------------------------------------------
// Max over nsample=32 into the multi-scale concat slot of l_feats[i+1].
// ---------------------------------------------------------------------------
__global__ void pool_concat_kernel(const float* __restrict__ act,
                                   float* __restrict__ lf,
                                   int C, int Cc, int coff, int total) {
    int t = blockIdx.x * blockDim.x + threadIdx.x;   // (b*np, c)
    if (t >= total) return;
    int p = t / C, c = t % C;
    float v = -3.0e38f;
    const float* a = act + ((size_t)p * 32) * C + c;
    for (int k = 0; k < 32; ++k) v = fmaxf(v, a[(size_t)k * C]);
    lf[(size_t)p * Cc + coff + c] = v;
}

// ---------------------------------------------------------------------------
// Three-NN + inverse-distance weights.
// ---------------------------------------------------------------------------
__global__ void nn3_kernel(const float* __restrict__ unknown,
                           const float* __restrict__ known,
                           int S, int M,
                           int* __restrict__ nidx, float* __restrict__ nw,
                           int total) {
    int t = blockIdx.x * blockDim.x + threadIdx.x;
    if (t >= total) return;
    int b = t / S;
    const float* u = unknown + (size_t)t * 3;
    const float* k = known + (size_t)b * M * 3;
    const float ux = u[0], uy = u[1], uz = u[2];
    float d0 = 3e38f, d1 = 3e38f, d2 = 3e38f;
    int   i0 = 0, i1 = 0, i2 = 0;
    for (int m = 0; m < M; ++m) {
        const float dx = k[(size_t)m * 3 + 0] - ux;
        const float dy = k[(size_t)m * 3 + 1] - uy;
        const float dz = k[(size_t)m * 3 + 2] - uz;
        const float d  = dx * dx + dy * dy + dz * dz;
        if (d < d0)      { d2 = d1; i2 = i1; d1 = d0; i1 = i0; d0 = d; i0 = m; }
        else if (d < d1) { d2 = d1; i2 = i1; d1 = d;  i1 = m; }
        else if (d < d2) { d2 = d;  i2 = m; }
    }
    float w0 = 1.0f / (fmaxf(d0, 0.0f) + 1e-8f);
    float w1 = 1.0f / (fmaxf(d1, 0.0f) + 1e-8f);
    float w2 = 1.0f / (fmaxf(d2, 0.0f) + 1e-8f);
    const float ws = w0 + w1 + w2;
    nidx[(size_t)t * 3 + 0] = i0; nw[(size_t)t * 3 + 0] = w0 / ws;
    nidx[(size_t)t * 3 + 1] = i1; nw[(size_t)t * 3 + 1] = w1 / ws;
    nidx[(size_t)t * 3 + 2] = i2; nw[(size_t)t * 3 + 2] = w2 / ws;
}

// A[row][c] = bf16( c<C2 ? sum_k w_k*f2[idx_k][c] : f1[row][c-C2] )
__global__ void interp_concat_kernel(const float* __restrict__ f2,
                                     const float* __restrict__ f1,
                                     const int* __restrict__ nidx,
                                     const float* __restrict__ nw,
                                     unsigned short* __restrict__ A,
                                     int S, int M, int C2, int C1,
                                     size_t total) {
    size_t t = (size_t)blockIdx.x * blockDim.x + threadIdx.x;
    if (t >= total) return;
    const int Kin = C1 + C2;
    const size_t row = t / Kin;
    const int    c   = (int)(t % Kin);
    float v;
    if (c < C2) {
        const int b = (int)(row / S);
        const int* id = nidx + row * 3;
        const float* w = nw + row * 3;
        const float* base = f2 + (size_t)b * M * C2;
        v = w[0] * base[(size_t)id[0] * C2 + c] +
            w[1] * base[(size_t)id[1] * C2 + c] +
            w[2] * base[(size_t)id[2] * C2 + c];
    } else {
        v = f1[row * C1 + (c - C2)];
    }
    A[t] = f2bf(v);
}

// (B, P, C) -> (B, C, P)
__global__ void transpose_out_kernel(const float* __restrict__ src,
                                     float* __restrict__ dst,
                                     int P, int C, int total) {
    int t = blockIdx.x * blockDim.x + threadIdx.x;
    if (t >= total) return;
    int b   = t / (P * C);
    int rem = t % (P * C);
    int p   = rem / C;
    int c   = rem % C;
    dst[((size_t)b * C + c) * P + p] = src[t];
}

// ---------------------------------------------------------------------------
// Host orchestration
// ---------------------------------------------------------------------------
static inline int ceil_div(int a, int b) { return (a + b - 1) / b; }
static inline int pad32(int x) { return (x + 31) & ~31; }

extern "C" void kernel_launch(void* const* d_in, const int* in_sizes, int n_in,
                              void* d_out, int out_size, void* d_ws, size_t ws_size,
                              hipStream_t stream) {
    (void)in_sizes; (void)n_in; (void)out_size; (void)ws_size;

    const int Bn = 2, N0 = 16384;
    const int NPOINT[4] = {4096, 1024, 256, 64};
    const float RAD[4][2] = {{0.05f, 0.1f}, {0.1f, 0.2f}, {0.2f, 0.4f}, {0.4f, 0.8f}};
    const int SA[4][4]  = {{3, 32, 32, 64}, {131, 64, 64, 128},
                           {259, 128, 128, 256}, {515, 256, 256, 512}};
    const int FPCH[4][4] = {{128, 128, 128, 128}, {384, 256, 128, 0},
                            {768, 256, 256, 0}, {1536, 512, 512, 0}};
    const int FPNL[4] = {3, 2, 2, 2};
    const int SPTS[5] = {16384, 4096, 1024, 256, 64};
    const int CCH[5]  = {0, 128, 256, 512, 1024};   // channels of l_feats[k]

    const float* pc = (const float*)d_in[0];

    // ---- deterministic bump allocator over d_ws ----
    char* wsb = (char*)d_ws;
    size_t off = 0;
    auto alloc = [&](size_t bytes) -> char* {
        off = (off + 255) & ~(size_t)255;
        char* p = wsb + off;
        off += bytes;
        return p;
    };

    unsigned short* WT  = (unsigned short*)alloc((size_t)16 << 20);      // bf16 weights
    float* DIST = (float*)alloc((size_t)Bn * N0 * 4);
    int*   IDX  = (int*)  alloc((size_t)Bn * 4096 * 4);
    int*   GI   = (int*)  alloc((size_t)Bn * 4096 * 32 * 4);
    const float* LX[5];
    LX[0] = pc;
    for (int i = 0; i < 4; ++i)
        LX[i + 1] = (const float*)alloc((size_t)Bn * NPOINT[i] * 3 * 4);
    float* LF[5] = {nullptr, nullptr, nullptr, nullptr, nullptr};
    for (int k = 1; k <= 4; ++k)
        LF[k] = (float*)alloc((size_t)Bn * SPTS[k] * CCH[k] * 4);
    float* LF0  = (float*)alloc((size_t)Bn * N0 * 128 * 4);
    int*   NNI  = (int*)  alloc((size_t)Bn * N0 * 3 * 4);
    float* NNW  = (float*)alloc((size_t)Bn * N0 * 3 * 4);
    unsigned short* ABUF = (unsigned short*)alloc((size_t)24 << 20);
    float* ACT0 = (float*)alloc((size_t)68 << 20);
    float* ACT1 = (float*)alloc((size_t)36 << 20);

    // ---- input pointer table (setup_inputs dict order) ----
    const float *sa_W[4][2][3], *sa_b[4][2][3];
    const float *fp_W[4][3],    *fp_b[4][3];
    int ai = 1;
    for (int i = 0; i < 4; ++i)
        for (int s = 0; s < 2; ++s)
            for (int j = 0; j < 3; ++j) {
                sa_W[i][s][j] = (const float*)d_in[ai++];
                sa_b[i][s][j] = (const float*)d_in[ai++];
            }
    for (int i = 0; i < 4; ++i)
        for (int j = 0; j < FPNL[i]; ++j) {
            fp_W[i][j] = (const float*)d_in[ai++];
            fp_b[i][j] = (const float*)d_in[ai++];
        }

    // ---- convert weights: W[ci][co] f32 -> Wt[co][Kpad] bf16 ----
    unsigned short* sa_Wt[4][2][3];
    unsigned short* fp_Wt[4][3];
    size_t wto = 0;
    for (int i = 0; i < 4; ++i)
        for (int s = 0; s < 2; ++s)
            for (int j = 0; j < 3; ++j) {
                const int ci = SA[i][j], co = SA[i][j + 1];
                const int Kp = (j == 0) ? pad32(ci) : ci;
                sa_Wt[i][s][j] = WT + wto; wto += (size_t)co * Kp;
                const int tot = co * Kp;
                cvt_weight_kernel<<<ceil_div(tot, 256), 256, 0, stream>>>(
                    sa_W[i][s][j], sa_Wt[i][s][j], ci, co, Kp, tot);
            }
    for (int i = 0; i < 4; ++i)
        for (int j = 0; j < FPNL[i]; ++j) {
            const int ci = FPCH[i][j], co = FPCH[i][j + 1];
            const int Kp = ci;                       // all fp ci are mult of 32
            fp_Wt[i][j] = WT + wto; wto += (size_t)co * Kp;
            const int tot = co * Kp;
            cvt_weight_kernel<<<ceil_div(tot, 256), 256, 0, stream>>>(
                fp_W[i][j], fp_Wt[i][j], ci, co, Kp, tot);
        }

    auto gemm = [&](const unsigned short* Aa, const unsigned short* Wtt,
                    const float* bb, float* oo, int M, int Nc, int K) {
        dim3 grid((unsigned)ceil_div(M / 32, 4), (unsigned)(Nc / 32));
        dim3 block(32, 4);
        gemm_bias_relu_kernel<<<grid, block, 0, stream>>>(Aa, Wtt, bb, oo, M, Nc, K);
    };
    auto cvt = [&](const float* s, unsigned short* d, size_t n) {
        cvt_act_kernel<<<(unsigned)((n + 255) / 256), 256, 0, stream>>>(s, d, n);
    };

    // =========================== SA levels ===========================
    int curN = N0;
    const float* cx = pc;
    const float* cf = nullptr;
    int Cf = 0;
    for (int i = 0; i < 4; ++i) {
        const int np = NPOINT[i];
        fps_kernel<<<Bn, 1024, 0, stream>>>(cx, curN, np, IDX, DIST);
        const int tg = Bn * np;
        gather_xyz_kernel<<<ceil_div(tg, 256), 256, 0, stream>>>(
            cx, IDX, (float*)LX[i + 1], np, curN, tg);

        const int c1 = SA[i][1], c2 = SA[i][2], c3 = SA[i][3];
        const int Kp0 = pad32(SA[i][0]);
        const int rows = Bn * np * 32;
        const int Cc = 2 * c3;
        for (int s = 0; s < 2; ++s) {
            const float r = RAD[i][s];
            ball_query_kernel<<<ceil_div(tg, 256), 256, 0, stream>>>(
                cx, LX[i + 1], curN, np, r * r, GI, tg);
            group_kernel<<<ceil_div(rows, 256), 256, 0, stream>>>(
                cx, cf, LX[i + 1], GI, ABUF, np, curN, Cf, Kp0, rows);

            gemm(ABUF, sa_Wt[i][s][0], sa_b[i][s][0], ACT0, rows, c1, Kp0);
            cvt(ACT0, ABUF, (size_t)rows * c1);
            gemm(ABUF, sa_Wt[i][s][1], sa_b[i][s][1], ACT1, rows, c2, c1);
            cvt(ACT1, ABUF, (size_t)rows * c2);
            gemm(ABUF, sa_Wt[i][s][2], sa_b[i][s][2], ACT0, rows, c3, c2);

            const int ptot = Bn * np * c3;
            pool_concat_kernel<<<ceil_div(ptot, 256), 256, 0, stream>>>(
                ACT0, LF[i + 1], c3, Cc, s * c3, ptot);
        }
        cx = LX[i + 1];
        cf = LF[i + 1];
        Cf = Cc;
        curN = np;
    }

    // =========================== FP levels ===========================
    for (int t = 0; t < 4; ++t) {
        const int lev = 3 - t;                        // uses params['fp'][lev]
        const int S = SPTS[lev], M = SPTS[lev + 1];
        const float* unk = LX[lev];
        const float* kn  = LX[lev + 1];
        const float* f2  = LF[lev + 1];
        const int C2 = CCH[lev + 1];
        const float* f1  = (lev == 0) ? nullptr : LF[lev];
        const int C1 = (lev == 0) ? 0 : CCH[lev];
        const int rows = Bn * S;
        const int Kin = C1 + C2;

        nn3_kernel<<<ceil_div(rows, 128), 128, 0, stream>>>(
            unk, kn, S, M, NNI, NNW, rows);
        const size_t itot = (size_t)rows * Kin;
        interp_concat_kernel<<<(unsigned)((itot + 255) / 256), 256, 0, stream>>>(
            f2, f1, NNI, NNW, ABUF, S, M, C2, C1, itot);

        const int nl = FPNL[lev];
        float* dst_final = (lev == 0) ? LF0 : LF[lev];
        int Kcur = Kin;
        for (int j = 0; j < nl; ++j) {
            const int co = FPCH[lev][j + 1];
            float* outp = (j == nl - 1) ? dst_final : ((j & 1) ? ACT1 : ACT0);
            gemm(ABUF, fp_Wt[lev][j], fp_b[lev][j], outp, rows, co, Kcur);
            if (j != nl - 1) cvt(outp, ABUF, (size_t)rows * co);
            Kcur = co;
        }
    }

    // =========================== Outputs ===========================
    float* out0 = (float*)d_out;                           // (B,128,16384)
    float* out1 = out0 + (size_t)Bn * 128 * N0;            // (B,1024,64)
    {
        const int tot0 = Bn * N0 * 128;
        transpose_out_kernel<<<ceil_div(tot0, 256), 256, 0, stream>>>(
            LF0, out0, N0, 128, tot0);
        const int tot1 = Bn * 64 * 1024;
        transpose_out_kernel<<<ceil_div(tot1, 256), 256, 0, stream>>>(
            LF[4], out1, 64, 1024, tot1);
    }
}
